// BCNet_44341242364144
// MI455X (gfx1250) — matrix operand order repaired and checked by reading, not compiled, not run on hardware
//
#include <hip/hip_runtime.h>

// ---------------------------------------------------------------------------
// BCNet bilinear attention logits for MI455X (gfx1250, wave32, WMMA).
//   v_ = relu(v @ Wv^T + bv)   (B*NV, HK)  -> bf16 workspace
//   q_ = relu(q @ Wq^T + bq)   (B*NQ, HK)  -> bf16 workspace
//   logits[b,h] = (v_[b] * h_mat[h]) @ q_[b]^T + h_bias[h]
// V_WMMA_F32_16X16X32_BF16, double-buffered LDS, 1 barrier per k-step,
// async global->LDS (ASYNCcnt) staging for the Q tile in kernel 3.
// ---------------------------------------------------------------------------

#define B_    32
#define NV_   512
#define NQ_   128
#define VDIM_ 2048
#define QDIM_ 1024
#define HK_   1536
#define HOUT_ 8

#define BM 128
#define BN 128
#define BK 32
#define LDSS 48   // LDS row stride in bf16 elements = 96 bytes (16B aligned, bank-skewed)

typedef float          v4f   __attribute__((ext_vector_type(4)));
typedef float          v8f   __attribute__((ext_vector_type(8)));
typedef __bf16         v4bf  __attribute__((ext_vector_type(4)));
typedef __bf16         v16bf __attribute__((ext_vector_type(16)));
typedef unsigned int   v4u   __attribute__((ext_vector_type(4)));
typedef unsigned short v4us  __attribute__((ext_vector_type(4)));

union FragCast  { v4u q[2]; v16bf bf; };
union PackCast  { v4bf bf; v4us us; };

__device__ __forceinline__ v4us cvt4(v4f x) {
  PackCast p;
  p.bf = __builtin_convertvector(x, v4bf);   // packed f32->bf16 (RNE)
  return p.us;
}
__device__ __forceinline__ unsigned short f2bf(float f) {
  unsigned int u = __float_as_uint(f);
  u += 0x7FFFu + ((u >> 16) & 1u);
  return (unsigned short)(u >> 16);
}

// ---- staging: per-thread half-row (r = tid>>1, 16 elements at kb) ----------

// fp32 global tile -> 16 bf16 held in registers (4x v4us)
__device__ __forceinline__ void stage_load_f32(const float* __restrict__ src, int ld,
                                               int tid, v4us s[4]) {
  const int r  = tid >> 1;
  const int kb = (tid & 1) << 4;
  const float* p = src + (size_t)r * ld + kb;
#pragma unroll
  for (int j = 0; j < 4; ++j)
    s[j] = cvt4(*(const v4f*)(p + 4 * j));
}

// bf16 global tile scaled per-column by fp32 scale[k] -> registers
__device__ __forceinline__ void stage_load_bf16_scaled(const unsigned short* __restrict__ src,
                                                       int ld, const float* __restrict__ scale,
                                                       int tid, v4us s[4]) {
  const int r  = tid >> 1;
  const int kb = (tid & 1) << 4;
  const unsigned short* p = src + (size_t)r * ld + kb;
  const v4u raw0 = *(const v4u*)(p);       // bf16 pairs k..k+7
  const v4u raw1 = *(const v4u*)(p + 8);   // bf16 pairs k+8..k+15
#pragma unroll
  for (int c = 0; c < 2; ++c) {
    const v4u raw = c ? raw1 : raw0;
#pragma unroll
    for (int half = 0; half < 2; ++half) {
      v4f f, sc;
#pragma unroll
      for (int t = 0; t < 2; ++t) {
        const unsigned int u = raw[2 * half + t];
        f[2 * t]     = __uint_as_float(u << 16);           // low bf16
        f[2 * t + 1] = __uint_as_float(u & 0xFFFF0000u);   // high bf16
        const int k = kb + 8 * c + 4 * half + 2 * t;
        sc[2 * t]     = scale[k];
        sc[2 * t + 1] = scale[k + 1];
      }
      s[2 * c + half] = cvt4(f * sc);
    }
  }
}

__device__ __forceinline__ void stage_store(unsigned short* __restrict__ dst, int tid,
                                            const v4us s[4]) {
  const int r  = tid >> 1;
  const int kb = (tid & 1) << 4;
  unsigned short* d = dst + r * LDSS + kb;
#pragma unroll
  for (int j = 0; j < 4; ++j)
    *(v4us*)(d + 4 * j) = s[j];
}

// Async global->LDS staging of a 128x32 bf16 tile (2x b128 per thread).
// LDS byte address comes from the vdst VGPR; low 32 bits of a generic LDS pointer are the
// workgroup-relative LDS byte offset, matching the ds_* addressing convention.
__device__ __forceinline__ void async_stage_bf16(const unsigned short* __restrict__ src, int ld,
                                                 unsigned short* __restrict__ lds, int tid) {
  const int r  = tid >> 1;
  const int kb = (tid & 1) << 4;
  const unsigned short* p = src + (size_t)r * ld + kb;
  const unsigned int l = (unsigned int)(uintptr_t)(lds + r * LDSS + kb);
  asm volatile("global_load_async_to_lds_b128 %0, %1, off"
               :: "v"(l), "v"(p) : "memory");
  asm volatile("global_load_async_to_lds_b128 %0, %1, off"
               :: "v"(l + 16u), "v"(p + 8) : "memory");
}
__device__ __forceinline__ void wait_async0() {
  asm volatile("s_wait_asynccnt 0" ::: "memory");
}

// ---- fragment loads (ISA VGPR layouts, 2x ds_load_b128 each) ---------------

// A (16x32 MxK): lane -> M=lane%16; lanes<16: K{0-7,16-23}; lanes>=16: K{8-15,24-31}
__device__ __forceinline__ v16bf load_afrag(const unsigned short* As, int mbase, int lane) {
  const unsigned short* p = As + (mbase + (lane & 15)) * LDSS + ((lane >> 4) << 3);
  FragCast f;
  f.q[0] = *(const v4u*)(p);
  f.q[1] = *(const v4u*)(p + 16);
  return f.bf;
}
// B (32x16 KxN): lane -> N=lane%16; lanes<16: K{0-7,16-23}; lanes>=16: K{8-15,24-31}
__device__ __forceinline__ v16bf load_bfrag(const unsigned short* Bs, int nbase, int lane) {
  const unsigned short* p = Bs + (nbase + (lane & 15)) * LDSS + ((lane >> 4) << 3);
  FragCast f;
  f.q[0] = *(const v4u*)(p);
  f.q[1] = *(const v4u*)(p + 16);
  return f.bf;
}

// ---------------------------------------------------------------------------
// Kernel 1/2: Y_bf16[M,N] = relu(X[M,K] . W[N,K]^T + bias[N])
// ---------------------------------------------------------------------------
__global__ __launch_bounds__(256)
void proj_relu_bf16(const float* __restrict__ X, const float* __restrict__ W,
                    const float* __restrict__ bias, unsigned short* __restrict__ Y,
                    int M, int N, int Kdim) {
  __shared__ unsigned short As[2][BM * LDSS];
  __shared__ unsigned short Bs[2][BN * LDSS];
  const int tid  = threadIdx.x;
  const int lane = tid & 31;
  const int wave = tid >> 5;
  const int wm   = (wave & 3) * 32;
  const int wn   = (wave >> 2) * 64;
  const int m0   = blockIdx.y * BM;
  const int n0   = blockIdx.x * BN;

  v8f acc[2][4];
#pragma unroll
  for (int i = 0; i < 2; ++i)
#pragma unroll
    for (int j = 0; j < 4; ++j)
      acc[i][j] = (v8f){0.f,0.f,0.f,0.f,0.f,0.f,0.f,0.f};

  const float* Xb = X + (size_t)m0 * Kdim;
  const float* Wb = W + (size_t)n0 * Kdim;

  v4us sa[4], sb[4];
  stage_load_f32(Xb, Kdim, tid, sa);
  stage_load_f32(Wb, Kdim, tid, sb);
  stage_store(As[0], tid, sa);
  stage_store(Bs[0], tid, sb);
  __syncthreads();

  const int steps = Kdim / BK;
  for (int s = 0; s < steps; ++s) {
    const int cur = s & 1;
    const bool more = (s + 1) < steps;
    if (more) {                       // issue next-tile global loads early
      stage_load_f32(Xb + (s + 1) * BK, Kdim, tid, sa);
      stage_load_f32(Wb + (s + 1) * BK, Kdim, tid, sb);
    }
    v16bf a[2], b[4];
#pragma unroll
    for (int i = 0; i < 2; ++i) a[i] = load_afrag(As[cur], wm + i * 16, lane);
#pragma unroll
    for (int j = 0; j < 4; ++j) b[j] = load_bfrag(Bs[cur], wn + j * 16, lane);
#pragma unroll
    for (int i = 0; i < 2; ++i)
#pragma unroll
      for (int j = 0; j < 4; ++j)
        acc[i][j] = __builtin_amdgcn_wmma_f32_16x16x32_bf16(
            false, a[i], false, b[j], (short)0, acc[i][j], false, false);
    if (more) {                       // stage into the other buffer
      stage_store(As[cur ^ 1], tid, sa);
      stage_store(Bs[cur ^ 1], tid, sb);
    }
    __syncthreads();
  }

  // Epilogue: +bias, relu, f32->bf16. C/D: vgpr r -> M=r+8*(lane>=16), N=lane%16.
  const int mlane = lane & 15;
  const int half  = lane >> 4;
#pragma unroll
  for (int i = 0; i < 2; ++i) {
#pragma unroll
    for (int j = 0; j < 4; ++j) {
      const int ncol = n0 + wn + j * 16 + mlane;
      const float bb = bias[ncol];
#pragma unroll
      for (int r = 0; r < 8; ++r) {
        const int mrow = m0 + wm + i * 16 + half * 8 + r;
        float val = acc[i][j][r] + bb;
        val = val > 0.f ? val : 0.f;
        Y[(size_t)mrow * N + ncol] = f2bf(val);
      }
    }
  }
}

// ---------------------------------------------------------------------------
// Kernel 3: per (b,h): out[b,h] = (v_[b] * h_mat[h]) @ q_[b]^T + h_bias[h]
// M=NV=512 (4 m-blocks), N=NQ=128 (1 n-block), K=HK=1536.
// A tile: register-staged (needs h_mat scaling); B tile: async global->LDS.
// ---------------------------------------------------------------------------
__global__ __launch_bounds__(256)
void bilinear_logits(const unsigned short* __restrict__ Vh,  // (B*NV, HK) bf16
                     const unsigned short* __restrict__ Qh,  // (B*NQ, HK) bf16
                     const float* __restrict__ h_mat,        // (HOUT, HK)
                     const float* __restrict__ h_bias,       // (HOUT)
                     float* __restrict__ out) {              // (B, HOUT, NV, NQ)
  __shared__ unsigned short As[2][BM * LDSS];
  __shared__ unsigned short Bs[2][BN * LDSS];
  const int tid  = threadIdx.x;
  const int lane = tid & 31;
  const int wave = tid >> 5;
  const int wm   = (wave & 3) * 32;
  const int wn   = (wave >> 2) * 64;
  const int bh   = blockIdx.z;
  const int b    = bh >> 3;
  const int h    = bh & 7;
  const int m0   = blockIdx.y * BM;   // within NV

  const unsigned short* Vb = Vh + (size_t)b * NV_ * HK_ + (size_t)m0 * HK_;
  const unsigned short* Qb = Qh + (size_t)b * NQ_ * HK_;
  const float* hm = h_mat + (size_t)h * HK_;

  v8f acc[2][4];
#pragma unroll
  for (int i = 0; i < 2; ++i)
#pragma unroll
    for (int j = 0; j < 4; ++j)
      acc[i][j] = (v8f){0.f,0.f,0.f,0.f,0.f,0.f,0.f,0.f};

  v4us sa[4];
  stage_load_bf16_scaled(Vb, HK_, hm, tid, sa);
  async_stage_bf16(Qb, HK_, Bs[0], tid);        // Q tile s=0: async -> LDS
  stage_store(As[0], tid, sa);
  wait_async0();
  __syncthreads();

  const int steps = HK_ / BK;
  for (int s = 0; s < steps; ++s) {
    const int cur = s & 1;
    const bool more = (s + 1) < steps;
    if (more) {
      async_stage_bf16(Qb + (s + 1) * BK, HK_, Bs[cur ^ 1], tid);   // ASYNCcnt
      stage_load_bf16_scaled(Vb + (s + 1) * BK, HK_, hm + (s + 1) * BK, tid, sa);
    }
    v16bf a[2], bb[4];
#pragma unroll
    for (int i = 0; i < 2; ++i) a[i] = load_afrag(As[cur], wm + i * 16, lane);
#pragma unroll
    for (int j = 0; j < 4; ++j) bb[j] = load_bfrag(Bs[cur], wn + j * 16, lane);
#pragma unroll
    for (int i = 0; i < 2; ++i)
#pragma unroll
      for (int j = 0; j < 4; ++j)
        acc[i][j] = __builtin_amdgcn_wmma_f32_16x16x32_bf16(
            false, a[i], false, bb[j], (short)0, acc[i][j], false, false);
    if (more) stage_store(As[cur ^ 1], tid, sa);
    wait_async0();                    // our async writes to Bs[next] are done
    __syncthreads();                  // everyone's writes visible
  }

  const float hb  = h_bias[h];
  const int mlane = lane & 15;
  const int half  = lane >> 4;
  float* outb = out + (size_t)bh * NV_ * NQ_;
#pragma unroll
  for (int i = 0; i < 2; ++i) {
#pragma unroll
    for (int j = 0; j < 4; ++j) {
      const int ncol = wn + j * 16 + mlane;   // n0 == 0 (NQ == BN)
#pragma unroll
      for (int r = 0; r < 8; ++r) {
        const int mrow = m0 + wm + i * 16 + half * 8 + r;
        outb[(size_t)mrow * NQ_ + ncol] = acc[i][j][r] + hb;
      }
    }
  }
}

// ---------------------------------------------------------------------------
extern "C" void kernel_launch(void* const* d_in, const int* in_sizes, int n_in,
                              void* d_out, int out_size, void* d_ws, size_t ws_size,
                              hipStream_t stream) {
  (void)in_sizes; (void)n_in; (void)out_size; (void)ws_size;
  const float* v     = (const float*)d_in[0];
  const float* q     = (const float*)d_in[1];
  const float* Wv    = (const float*)d_in[2];
  const float* bv    = (const float*)d_in[3];
  const float* Wq    = (const float*)d_in[4];
  const float* bq    = (const float*)d_in[5];
  const float* hmat  = (const float*)d_in[6];
  const float* hbias = (const float*)d_in[7];
  float* out = (float*)d_out;

  unsigned short* vws = (unsigned short*)d_ws;                       // (B*NV, HK) bf16
  unsigned short* qws = (unsigned short*)((char*)d_ws +
                         (size_t)B_ * NV_ * HK_ * sizeof(unsigned short)); // (B*NQ, HK)

  dim3 blk(256);
  // v_ = relu(v @ Wv^T + bv): M=16384, N=1536, K=2048
  proj_relu_bf16<<<dim3(HK_ / BN, (B_ * NV_) / BM), blk, 0, stream>>>(
      v, Wv, bv, vws, B_ * NV_, HK_, VDIM_);
  // q_ = relu(q @ Wq^T + bq): M=4096, N=1536, K=1024
  proj_relu_bf16<<<dim3(HK_ / BN, (B_ * NQ_) / BM), blk, 0, stream>>>(
      q, Wq, bq, qws, B_ * NQ_, HK_, QDIM_);
  // logits: per (b,h) 512x128x1536 GEMM
  bilinear_logits<<<dim3(1, NV_ / BM, B_ * HOUT_), blk, 0, stream>>>(
      vws, qws, hmat, hbias, out);
}
